// AttentionDecoder_80934363725951
// MI455X (gfx1250) — matrix operand order repaired and checked by reading, not compiled, not run on hardware
//
#include <hip/hip_runtime.h>
#include <hip/hip_bf16.h>
#include <stddef.h>

// ---------------- constants ----------------
#define BATCH   1024
#define HID     256
#define K2      512     // 2*HID
#define LSEQ    300     // actual encoder length
#define LPAD    320     // ENC_MAX_LEN
#define VOCAB   81
#define TSTEPS  150
#define G4      1024    // 4*HID

// GEMM tiling
#define BM      128     // block M tile (8 waves x 16)
#define BN      64      // block N tile
#define KC      64      // K chunk staged in LDS
#define BSTRIDE 72      // LDS row stride in shorts (144B: 16B aligned, bank-spread)

typedef __attribute__((ext_vector_type(16))) __bf16 v16bf;
typedef __attribute__((ext_vector_type(8)))  float  v8f;

union FragBf { uint4 q[2]; unsigned u[8]; v16bf v; };

__device__ __forceinline__ unsigned short f2bf(float x) {
    unsigned u = __float_as_uint(x);
    u = (u + 0x7FFFu + ((u >> 16) & 1u)) >> 16;   // round-to-nearest-even
    return (unsigned short)u;
}
__device__ __forceinline__ float sigm(float x) { return 1.0f / (1.0f + __expf(-x)); }

// ---------------- prep kernels ----------------
__global__ void k_cvt_bf16(const float* __restrict__ src, unsigned short* __restrict__ dst, int n) {
    int i = blockIdx.x * blockDim.x + threadIdx.x;
    if (i < n) dst[i] = f2bf(src[i]);
}

// Build W_ihh = [W_ih | W_hh] (rows 4H=1024, cols 512) in bf16; bias_ihh = b_ih + b_hh
__global__ void k_build_wihh(const float* __restrict__ Wih, const float* __restrict__ Whh,
                             const float* __restrict__ bih, const float* __restrict__ bhh,
                             unsigned short* __restrict__ Wihh, float* __restrict__ bIhh) {
    int i = blockIdx.x * blockDim.x + threadIdx.x;      // over 1024*512
    if (i >= G4 * K2) return;
    int n = i / K2, k = i % K2;
    float v = (k < HID) ? Wih[n * HID + k] : Whh[n * HID + (k - HID)];
    Wihh[i] = f2bf(v);
    if (k == 0) bIhh[n] = bih[n] + bhh[n];
}

// pe_row (the reference's single-row positional quirk: uses L=300 as the "position")
__global__ void k_init_pe(float* __restrict__ pe) {
    int i = threadIdx.x;                                 // 0..127
    if (i < HID / 2) {
        float div = __expf((2.0f * (float)i) * (-logf(10000.0f) / (float)HID)) * (float)LSEQ;
        pe[2 * i]     = __sinf(div);
        pe[2 * i + 1] = __cosf(div);
    }
}

__global__ void k_init_state(float* __restrict__ h, float* __restrict__ c,
                             int* __restrict__ xd, float* __restrict__ lossAcc) {
    int i = blockIdx.x * blockDim.x + threadIdx.x;       // over 1024*256
    if (i < BATCH * HID) { h[i] = 0.0f; c[i] = 0.0f; }
    if (i < BATCH) xd[i] = 1;                            // SOS_IDX
    if (i == 0) *lossAcc = 0.0f;
}

// ---------------- per-step elementwise ----------------
// att_in = [xe|h] (bf16), comb_in[:,0:256] = xe (bf16), cat_xh[:,256:512] = h (bf16)
__global__ void k_build_attin(const int* __restrict__ xd, const float* __restrict__ emb,
                              const float* __restrict__ h,
                              unsigned short* __restrict__ attIn,
                              unsigned short* __restrict__ combIn,
                              unsigned short* __restrict__ catXh) {
    int i = blockIdx.x * blockDim.x + threadIdx.x;       // over 1024*256
    if (i >= BATCH * HID) return;
    int b = i >> 8, hh = i & 255;
    float xe = emb[xd[b] * HID + hh];
    unsigned short xeb = f2bf(xe);
    unsigned short hb  = f2bf(h[i]);
    attIn[(size_t)b * K2 + hh]       = xeb;
    attIn[(size_t)b * K2 + HID + hh] = hb;
    combIn[(size_t)b * K2 + hh]      = xeb;
    catXh[(size_t)b * K2 + HID + hh] = hb;
}

// ---------------- generic bf16 WMMA GEMM: C[M,N] = A[M,K] * W[N,K]^T + bias ----------------
// grid = (N/64, M/128), block = 256 (8 waves). Each wave: 16(M) x 64(N) strip.
// Weight tile (64 N-rows x 64 K) is staged in LDS once per K-chunk, shared by all 8 waves.
__global__ __launch_bounds__(256)
void k_gemm_bf16(const unsigned short* __restrict__ A, int lda,
                 const unsigned short* __restrict__ Bw, int ldb,
                 const float* __restrict__ bias,
                 float* __restrict__ Cf, int ldc,
                 unsigned short* __restrict__ Cb, int ldcb,
                 int K, int relu) {
    __shared__ unsigned short Bs[BN * BSTRIDE];          // 64 rows x 144B = 9216B

    const int tid  = threadIdx.x;
    const int lane = tid & 31;
    const int wv   = tid >> 5;
    const int mb   = blockIdx.y * BM + wv * 16;
    const int nb   = blockIdx.x * BN;
    const int half = lane >> 4;
    const int l15  = lane & 15;

    // cooperative B-tile staging assignments: 256 threads, 64 rows x 128B; 32B per thread
    const int srow = tid >> 2;                           // 0..63
    const int squt = tid & 3;                            // 0..3 (32B quarters)
    const unsigned short* gB0 = Bw + (size_t)(nb + srow) * ldb + squt * 16;
    uint4* lB = (uint4*)((char*)Bs + srow * (BSTRIDE * 2) + squt * 32);

    v8f acc[4] = {};
    const int mrow = mb + l15;
    const unsigned short* aRow = A + (size_t)mrow * lda;

    for (int k0 = 0; k0 < K; k0 += KC) {
        // stage weight tile chunk into LDS
        {
            const uint4* gp = (const uint4*)(gB0 + k0);
            uint4 w0 = gp[0];
            uint4 w1 = gp[1];
            lB[0] = w0;
            lB[1] = w1;
        }
        // prefetch next chunk of weights while we compute on this one
        if (k0 + KC < K) {
            __builtin_prefetch(gB0 + k0 + KC, 0, 3);
        }
        __syncthreads();

#pragma unroll
        for (int kk2 = 0; kk2 < KC; kk2 += 32) {
            // A fragment: issue global loads first so they overlap the LDS reads
            FragBf a;
#pragma unroll
            for (int v = 0; v < 8; ++v) {
                int kk = k0 + kk2 + ((v < 4) ? 0 : 16) + half * 8 + (v & 3) * 2;
                a.u[v] = *(const unsigned*)(aRow + kk);
            }
            // all four B fragments as one LDS clause, single wait
            FragBf bfr[4];
#pragma unroll
            for (int j = 0; j < 4; ++j) {
                const char* rowp = (const char*)Bs + (j * 16 + l15) * (BSTRIDE * 2)
                                 + kk2 * 2 + half * 16;
                bfr[j].q[0] = *(const uint4*)rowp;
                bfr[j].q[1] = *(const uint4*)(rowp + 32);
            }
            // WMMA burst on independent accumulators
#pragma unroll
            for (int j = 0; j < 4; ++j) {
                acc[j] = __builtin_amdgcn_wmma_f32_16x16x32_bf16(
                    false, a.v, false, bfr[j].v, (short)0, acc[j], false, false);
            }
        }
        __syncthreads();
    }

#pragma unroll
    for (int j = 0; j < 4; ++j) {
        int n = nb + j * 16 + l15;
        float bv = bias[n];
#pragma unroll
        for (int r = 0; r < 8; ++r) {
            int m = mb + r + 8 * half;
            float val = acc[j][r] + bv;
            if (relu) val = fmaxf(val, 0.0f);
            if (Cf) Cf[(size_t)m * ldc + n] = val;
            if (Cb) Cb[(size_t)m * ldcb + n] = f2bf(val);
        }
    }
}

// ---------------- softmax(320) + context via emb gather ----------------
// one block per batch row; 256 threads (thread owns hidden index tid)
__global__ __launch_bounds__(256)
void k_softmax_ctx(const float* __restrict__ scores, const int* __restrict__ encOut,
                   const float* __restrict__ emb, const float* __restrict__ pe,
                   unsigned short* __restrict__ combIn) {
    __shared__ float sw[LPAD];
    __shared__ int   sidx[LSEQ];
    __shared__ float red[256];
    const int b = blockIdx.x, tid = threadIdx.x;

    for (int i = tid; i < LPAD; i += 256) sw[i] = scores[(size_t)b * LPAD + i];
    for (int i = tid; i < LSEQ; i += 256) sidx[i] = encOut[(size_t)b * LSEQ + i];
    __syncthreads();

    // max over 320
    float m = sw[tid];
    if (tid < LPAD - 256) m = fmaxf(m, sw[256 + tid]);
    red[tid] = m; __syncthreads();
    for (int s = 128; s > 0; s >>= 1) { if (tid < s) red[tid] = fmaxf(red[tid], red[tid + s]); __syncthreads(); }
    float mx = red[0]; __syncthreads();

    // exp + sums
    float e1 = __expf(sw[tid] - mx);
    float e2 = 0.0f;
    if (tid < LPAD - 256) e2 = __expf(sw[256 + tid] - mx);
    sw[tid] = e1;
    if (tid < LPAD - 256) sw[256 + tid] = e2;
    red[tid] = e1 + e2; __syncthreads();
    for (int s = 128; s > 0; s >>= 1) { if (tid < s) red[tid] += red[tid + s]; __syncthreads(); }
    float ssum = red[0]; __syncthreads();
    red[tid] = (tid < LSEQ ? e1 : 0.0f) + ((256 + tid) < LSEQ ? e2 : 0.0f);
    __syncthreads();
    for (int s = 128; s > 0; s >>= 1) { if (tid < s) red[tid] += red[tid + s]; __syncthreads(); }
    float s300 = red[0]; __syncthreads();

    // context: sum_{l<300} e_l * emb[idx_l][tid]  + s300 * pe[tid], all / ssum
    float acc = 0.0f;
    for (int l = 0; l < LSEQ; ++l) {
        acc = fmaf(sw[l], emb[(size_t)sidx[l] * HID + tid], acc);
    }
    float a = (acc + s300 * pe[tid]) / ssum;
    combIn[(size_t)b * K2 + HID + tid] = f2bf(a);
}

// ---------------- LSTM pointwise (torch gate order i,f,g,o) ----------------
__global__ void k_lstm(const float* __restrict__ gates, float* __restrict__ h, float* __restrict__ c) {
    int i = blockIdx.x * blockDim.x + threadIdx.x;       // over 1024*256
    if (i >= BATCH * HID) return;
    int b = i >> 8, hh = i & 255;
    size_t base = (size_t)b * G4;
    float ig = gates[base + hh];
    float fg = gates[base + HID + hh];
    float gg = gates[base + 2 * HID + hh];
    float og = gates[base + 3 * HID + hh];
    float c2 = sigm(fg) * c[i] + sigm(ig) * tanhf(gg);
    float h2 = sigm(og) * tanhf(c2);
    c[i] = c2; h[i] = h2;
}

// ---------------- logits + log_softmax loss + greedy argmax ----------------
// one block per batch row; 128 threads; threads 0..80 own a vocab class
__global__ __launch_bounds__(128)
void k_logits(const float* __restrict__ h, const float* __restrict__ Wfc,
              const float* __restrict__ bfc, const int* __restrict__ tgt, int t,
              float* __restrict__ outLogits, int* __restrict__ xdNext,
              float* __restrict__ lossAcc) {
    __shared__ float hrow[HID];
    __shared__ float llog[VOCAB];
    __shared__ float red[128];
    __shared__ int   redi[128];
    const int b = blockIdx.x, tid = threadIdx.x;

    hrow[tid] = h[(size_t)b * HID + tid];
    hrow[tid + 128] = h[(size_t)b * HID + tid + 128];
    __syncthreads();

    float logit = -3.0e38f;
    if (tid < VOCAB) {
        float acc = bfc[tid];
        const float* wr = Wfc + (size_t)tid * HID;
#pragma unroll 8
        for (int k = 0; k < HID; ++k) acc = fmaf(hrow[k], wr[k], acc);
        logit = acc;
        llog[tid] = acc;
        outLogits[(size_t)b * VOCAB + tid] = acc;
    }

    // argmax (first occurrence) + max
    red[tid] = logit; redi[tid] = tid; __syncthreads();
    for (int s = 64; s > 0; s >>= 1) {
        if (tid < s) {
            float ov = red[tid + s]; int oi = redi[tid + s];
            if (ov > red[tid] || (ov == red[tid] && oi < redi[tid])) { red[tid] = ov; redi[tid] = oi; }
        }
        __syncthreads();
    }
    float mx = red[0]; int amax = redi[0]; __syncthreads();

    red[tid] = (tid < VOCAB) ? __expf(logit - mx) : 0.0f; __syncthreads();
    for (int s = 64; s > 0; s >>= 1) { if (tid < s) red[tid] += red[tid + s]; __syncthreads(); }

    if (tid == 0) {
        int y = tgt[(size_t)b * TSTEPS + t];
        float loss = (mx + logf(red[0])) - llog[y];      // -(log_softmax[y])
        atomicAdd(lossAcc, loss);
        xdNext[b] = amax;
    }
}

__global__ void k_finalize(const float* __restrict__ lossAcc, float* __restrict__ out) {
    if (threadIdx.x == 0 && blockIdx.x == 0)
        out[(size_t)TSTEPS * BATCH * VOCAB] = *lossAcc / (float)(TSTEPS * BATCH);
}

// ---------------- host side ----------------
static inline size_t alignup(size_t x) { return (x + 255) & ~(size_t)255; }

extern "C" void kernel_launch(void* const* d_in, const int* in_sizes, int n_in,
                              void* d_out, int out_size, void* d_ws, size_t ws_size,
                              hipStream_t stream) {
    (void)in_sizes; (void)n_in; (void)out_size; (void)ws_size;
    const int*   enc_out = (const int*)d_in[0];
    const int*   tgt     = (const int*)d_in[1];
    const float* emb     = (const float*)d_in[3];
    const float* W_att   = (const float*)d_in[4];
    const float* b_att   = (const float*)d_in[5];
    const float* W_comb  = (const float*)d_in[6];
    const float* b_comb  = (const float*)d_in[7];
    const float* W_ih    = (const float*)d_in[8];
    const float* b_ih    = (const float*)d_in[9];
    const float* W_hh    = (const float*)d_in[10];
    const float* b_hh    = (const float*)d_in[11];
    const float* W_fc    = (const float*)d_in[12];
    const float* b_fc    = (const float*)d_in[13];
    float* out = (float*)d_out;

    char* ws = (char*)d_ws;
    size_t off = 0;
    auto alloc = [&](size_t bytes) { char* p = ws + off; off = alignup(off + bytes); return p; };

    unsigned short* wAtt  = (unsigned short*)alloc((size_t)LPAD * K2 * 2);
    unsigned short* wComb = (unsigned short*)alloc((size_t)HID * K2 * 2);
    unsigned short* wIhh  = (unsigned short*)alloc((size_t)G4 * K2 * 2);
    float* bIhh   = (float*)alloc(G4 * 4);
    float* pe     = (float*)alloc(HID * 4);
    unsigned short* attIn  = (unsigned short*)alloc((size_t)BATCH * K2 * 2);
    unsigned short* combIn = (unsigned short*)alloc((size_t)BATCH * K2 * 2);
    unsigned short* catXh  = (unsigned short*)alloc((size_t)BATCH * K2 * 2);
    float* scores = (float*)alloc((size_t)BATCH * LPAD * 4);
    float* gates  = (float*)alloc((size_t)BATCH * G4 * 4);
    float* hState = (float*)alloc((size_t)BATCH * HID * 4);
    float* cState = (float*)alloc((size_t)BATCH * HID * 4);
    int* xdA      = (int*)alloc(BATCH * 4);
    int* xdB      = (int*)alloc(BATCH * 4);
    float* lossAcc = (float*)alloc(4);

    // ---- one-time prep (per launch; deterministic) ----
    k_cvt_bf16<<<(LPAD * K2 + 255) / 256, 256, 0, stream>>>(W_att, wAtt, LPAD * K2);
    k_cvt_bf16<<<(HID * K2 + 255) / 256, 256, 0, stream>>>(W_comb, wComb, HID * K2);
    k_build_wihh<<<(G4 * K2 + 255) / 256, 256, 0, stream>>>(W_ih, W_hh, b_ih, b_hh, wIhh, bIhh);
    k_init_pe<<<1, 128, 0, stream>>>(pe);
    k_init_state<<<(BATCH * HID + 255) / 256, 256, 0, stream>>>(hState, cState, xdA, lossAcc);

    int* xdCur = xdA;
    int* xdNext = xdB;
    const int ew = (BATCH * HID + 255) / 256;

    for (int t = 0; t < TSTEPS; ++t) {
        k_build_attin<<<ew, 256, 0, stream>>>(xdCur, emb, hState, attIn, combIn, catXh);

        // scores = att_in @ W_att^T + b_att
        k_gemm_bf16<<<dim3(LPAD / BN, BATCH / BM), 256, 0, stream>>>(
            attIn, K2, wAtt, K2, b_att, scores, LPAD, (unsigned short*)nullptr, 0, K2, 0);

        k_softmax_ctx<<<BATCH, 256, 0, stream>>>(scores, enc_out, emb, pe, combIn);

        // x = relu(comb_in @ W_comb^T + b_comb) -> bf16 into cat_xh[:,0:256]
        k_gemm_bf16<<<dim3(HID / BN, BATCH / BM), 256, 0, stream>>>(
            combIn, K2, wComb, K2, b_comb, (float*)nullptr, 0, catXh, K2, K2, 1);

        // gates = [x|h] @ [W_ih|W_hh]^T + (b_ih+b_hh)
        k_gemm_bf16<<<dim3(G4 / BN, BATCH / BM), 256, 0, stream>>>(
            catXh, K2, wIhh, K2, bIhh, gates, G4, (unsigned short*)nullptr, 0, K2, 0);

        k_lstm<<<ew, 256, 0, stream>>>(gates, hState, cState);

        k_logits<<<BATCH, 128, 0, stream>>>(
            hState, W_fc, b_fc, tgt, t, out + (size_t)t * BATCH * VOCAB, xdNext, lossAcc);

        int* tmp = xdCur; xdCur = xdNext; xdNext = tmp;
    }

    k_finalize<<<1, 64, 0, stream>>>(lossAcc, out);
}
